// NystromAttention_83227876262211
// MI455X (gfx1250) — matrix operand order repaired and checked by reading, not compiled, not run on hardware
//
#include <hip/hip_runtime.h>
#include <hip/hip_bf16.h>

typedef __attribute__((ext_vector_type(16))) __bf16 v16bf;
typedef __attribute__((ext_vector_type(8)))  __bf16 bfx8;
typedef __attribute__((ext_vector_type(8)))  float  v8f;
typedef __attribute__((ext_vector_type(2)))  float  v2f;

#define DEV __device__ __forceinline__

// ---------------- LDS helpers ----------------
DEV unsigned lds_addr(const void* p) {
  // generic -> LDS(addrspace 3) -> 32-bit LDS byte offset
  return (unsigned)(unsigned long long)(__attribute__((address_space(3))) const void*)p;
}
DEV void async_b128(unsigned lds_off, const void* gaddr) {
  asm volatile("global_load_async_to_lds_b128 %0, %1, off"
               :: "v"(lds_off), "v"(gaddr) : "memory");
}
DEV void wait_async0() {
  asm volatile("s_wait_asynccnt 0x0" ::: "memory");
}

// ---------------- bf16 WMMA fragment loads (CDNA5 16x16x32 layouts) ----------------
// A (16x32, MxK) from row-major [*, lda]: lane<16 -> row=lane, K 0..7 & 16..23
//                                         lane>=16 -> row=lane-16, K 8..15 & 24..31
DEV v16bf load_frag_a(const __bf16* A, int lda, int row0, int k0, int lane) {
  int r = lane & 15, hi = lane >> 4;
  const __bf16* p = A + (size_t)(row0 + r) * lda + k0 + hi * 8;
  union { v16bf v; bfx8 h[2]; } u;
  u.h[0] = *(const bfx8*)p;
  u.h[1] = *(const bfx8*)(p + 16);
  return u.v;
}
// B (32x16, KxN) from B-transposed row-major [N, ldb]: lane<16 -> col=lane, K 0..7 & 16..23
//                                                      lane>=16 -> col=lane-16, K 8..15 & 24..31
DEV v16bf load_frag_b(const __bf16* BT, int ldb, int n0, int k0, int lane) {
  int n = lane & 15, hi = lane >> 4;
  const __bf16* p = BT + (size_t)(n0 + n) * ldb + k0 + hi * 8;
  union { v16bf v; bfx8 h[2]; } u;
  u.h[0] = *(const bfx8*)p;
  u.h[1] = *(const bfx8*)(p + 16);
  return u.v;
}

// Register-double-buffered 16(M)x64(N) tile straight from global (for batched GEMMs).
DEV void wave_tile_16x64(const __bf16* A, int lda, const __bf16* BT, int ldb,
                         int row0, int n0, int K, v8f acc[4]) {
  int lane = threadIdx.x & 31;
  v16bf a  = load_frag_a(A, lda, row0, 0, lane);
  v16bf b0 = load_frag_b(BT, ldb, n0,      0, lane);
  v16bf b1 = load_frag_b(BT, ldb, n0 + 16, 0, lane);
  v16bf b2 = load_frag_b(BT, ldb, n0 + 32, 0, lane);
  v16bf b3 = load_frag_b(BT, ldb, n0 + 48, 0, lane);
  for (int k0 = 0; k0 < K; k0 += 32) {
    v16bf an = a, c0 = b0, c1 = b1, c2 = b2, c3 = b3;
    if (k0 + 32 < K) {  // prefetch next k-step before issuing WMMAs
      an = load_frag_a(A, lda, row0, k0 + 32, lane);
      c0 = load_frag_b(BT, ldb, n0,      k0 + 32, lane);
      c1 = load_frag_b(BT, ldb, n0 + 16, k0 + 32, lane);
      c2 = load_frag_b(BT, ldb, n0 + 32, k0 + 32, lane);
      c3 = load_frag_b(BT, ldb, n0 + 48, k0 + 32, lane);
    }
    acc[0] = __builtin_amdgcn_wmma_f32_16x16x32_bf16(false, a, false, b0, (short)0, acc[0], false, false);
    acc[1] = __builtin_amdgcn_wmma_f32_16x16x32_bf16(false, a, false, b1, (short)0, acc[1], false, false);
    acc[2] = __builtin_amdgcn_wmma_f32_16x16x32_bf16(false, a, false, b2, (short)0, acc[2], false, false);
    acc[3] = __builtin_amdgcn_wmma_f32_16x16x32_bf16(false, a, false, b3, (short)0, acc[3], false, false);
    a = an; b0 = c0; b1 = c1; b2 = c2; b3 = c3;
  }
}

// ---------------- LDS-staged big GEMM core (128x64 tile, 256 threads, async copies) ----
// LDS tiles are chunk-swizzled: 16B chunk (row,q) lives at (q*ROWS + row)*16 bytes, so
// fragment ds_load_b128s are lane-consecutive (conflict-free).
DEV v16bf lds_frag_a_sw(const __bf16* sA, int row0, int lane) {  // ROWS=128
  int row = row0 + (lane & 15), hi = lane >> 4;
  union { v16bf v; bfx8 h[2]; } u;
  u.h[0] = *(const bfx8*)(sA + (size_t)(hi * 128 + row) * 8);
  u.h[1] = *(const bfx8*)(sA + (size_t)((hi + 2) * 128 + row) * 8);
  return u.v;
}
DEV v16bf lds_frag_b_sw(const __bf16* sB, int n0, int lane) {    // ROWS=64
  int n = n0 + (lane & 15), hi = lane >> 4;
  union { v16bf v; bfx8 h[2]; } u;
  u.h[0] = *(const bfx8*)(sB + (size_t)(hi * 64 + n) * 8);
  u.h[1] = *(const bfx8*)(sB + (size_t)((hi + 2) * 64 + n) * 8);
  return u.v;
}

DEV void async_copy_tiles(const __bf16* Ag, int lda, const __bf16* Bg, int ldb,
                          int row0blk, int n0blk, int k0,
                          __bf16* dA, __bf16* dB) {
  int tid = threadIdx.x;
  // A tile: 128 rows x 32 elems = 512 16B chunks; 2 per thread.
#pragma unroll
  for (int i = 0; i < 2; ++i) {
    int c = tid + i * 256;
    int row = c >> 2, q = c & 3;                      // q fastest -> coalesced global reads
    const __bf16* g = Ag + (size_t)(row0blk + row) * lda + k0 + q * 8;
    async_b128(lds_addr(dA + (size_t)(q * 128 + row) * 8), g);
  }
  // B tile: 64 rows x 32 elems = 256 16B chunks; 1 per thread.
  {
    int row = tid >> 2, q = tid & 3;
    const __bf16* g = Bg + (size_t)(n0blk + row) * ldb + k0 + q * 8;
    async_b128(lds_addr(dB + (size_t)(q * 64 + row) * 8), g);
  }
}

// Fills this wave's acc[4] for rows [row0blk + wave*16) and cols [n0blk, n0blk+64).
DEV void staged_acc(const __bf16* Ag, int lda, const __bf16* Bg, int ldb,
                    int row0blk, int n0blk, int K, v8f acc[4],
                    __bf16* smA, __bf16* smB) {  // smA: 2*4096 el, smB: 2*2048 el
  int wave = threadIdx.x >> 5, lane = threadIdx.x & 31;
  int KT = K >> 5;
  async_copy_tiles(Ag, lda, Bg, ldb, row0blk, n0blk, 0, smA, smB);
  for (int kt = 0; kt < KT; ++kt) {
    int cur = kt & 1;
    wait_async0();
    __syncthreads();  // tile (cur) visible to all waves
    if (kt + 1 < KT)  // overlap next copy with this tile's WMMAs
      async_copy_tiles(Ag, lda, Bg, ldb, row0blk, n0blk, (kt + 1) << 5,
                       smA + (cur ^ 1) * 4096, smB + (cur ^ 1) * 2048);
    v16bf a = lds_frag_a_sw(smA + cur * 4096, wave * 16, lane);
#pragma unroll
    for (int t = 0; t < 4; ++t) {
      v16bf b = lds_frag_b_sw(smB + cur * 2048, t * 16, lane);
      acc[t] = __builtin_amdgcn_wmma_f32_16x16x32_bf16(false, a, false, b, (short)0, acc[t], false, false);
    }
    __syncthreads();  // all waves done reading (cur) before it is overwritten
  }
}

// ---------------- conversion / transpose / landmarks ----------------
__global__ void f32_to_bf16(const float* __restrict__ in, __bf16* __restrict__ out, long long n) {
  long long i = (long long)blockIdx.x * blockDim.x + threadIdx.x;
  if (i < n) out[i] = (__bf16)in[i];
}

__global__ void transpose_f32_to_bf16(const float* __restrict__ in, __bf16* __restrict__ out,
                                      int R, int C) {  // in [R][C] -> out [C][R]
  long long i = (long long)blockIdx.x * blockDim.x + threadIdx.x;
  if (i < (long long)R * C) {
    int r = (int)(i / C), c = (int)(i % C);
    out[(size_t)c * R + r] = (__bf16)in[i];
  }
}

// q/k: [bh, 4096, 64]; landmarks = mean over 64-token groups -> [bh, 64, 64]
__global__ void landmark_means(const __bf16* __restrict__ qbf, const __bf16* __restrict__ kbf,
                               __bf16* __restrict__ ql, __bf16* __restrict__ kl) {
  int idx = blockIdx.x * blockDim.x + threadIdx.x;  // 262144 total
  int j = idx & 63, lm = (idx >> 6) & 63, bh = idx >> 12;
  size_t base = ((size_t)bh * 4096 + lm * 64) * 64 + j;
  float sq = 0.f, sk = 0.f;
  for (int t = 0; t < 64; ++t) {
    sq += (float)qbf[base + (size_t)t * 64];
    sk += (float)kbf[base + (size_t)t * 64];
  }
  ql[idx] = (__bf16)(sq * (1.f / 64.f));
  kl[idx] = (__bf16)(sk * (1.f / 64.f));
}

// ---------------- big staged GEMMs ----------------
// QKV projection: scatter epilogue (q scaled by 1/sqrt(dh), v stored transposed).
__global__ void __launch_bounds__(256) qkv_gemm_staged(const __bf16* __restrict__ Xbf,
                                                       const __bf16* __restrict__ WqkvT,
                                                       const float* __restrict__ bqkv,
                                                       __bf16* __restrict__ qbf,
                                                       __bf16* __restrict__ kbf,
                                                       __bf16* __restrict__ vT) {
  __shared__ __align__(16) __bf16 smA[2 * 4096];
  __shared__ __align__(16) __bf16 smB[2 * 2048];
  int wave = threadIdx.x >> 5, lane = threadIdx.x & 31;
  int row0blk = blockIdx.x * 128, n0 = blockIdx.y * 64;
  v8f acc[4] = {};
  staged_acc(Xbf, 1024, WqkvT, 1024, row0blk, n0, 1024, acc, smA, smB);
  int nn = lane & 15, hi = lane >> 4, row0 = row0blk + wave * 16;
#pragma unroll
  for (int t = 0; t < 4; ++t)
#pragma unroll
    for (int e = 0; e < 8; ++e) {
      int row = row0 + e + hi * 8;       // global token row in [0,16384)
      int col = n0 + t * 16 + nn;        // in [0,3072)
      float val = acc[t][e] + bqkv[col];
      int part = col >> 10, dcol = col & 1023;
      int head = dcol >> 6, j = dcol & 63;
      int bidx = row >> 12, tok = row & 4095;
      size_t bh = (size_t)bidx * 16 + head;
      if (part == 0)      qbf[(bh * 4096 + tok) * 64 + j] = (__bf16)(val * 0.125f);
      else if (part == 1) kbf[(bh * 4096 + tok) * 64 + j] = (__bf16)val;
      else                vT[(bh * 64 + j) * 4096 + tok]  = (__bf16)val;
    }
}

// Output projection: fp32 out + bias, straight into d_out.
__global__ void __launch_bounds__(256) final_gemm_staged(const __bf16* __restrict__ A,
                                                         const __bf16* __restrict__ WoutT,
                                                         const float* __restrict__ bout,
                                                         float* __restrict__ C) {
  __shared__ __align__(16) __bf16 smA[2 * 4096];
  __shared__ __align__(16) __bf16 smB[2 * 2048];
  int wave = threadIdx.x >> 5, lane = threadIdx.x & 31;
  int row0blk = blockIdx.x * 128, n0 = blockIdx.y * 64;
  v8f acc[4] = {};
  staged_acc(A, 1024, WoutT, 1024, row0blk, n0, 1024, acc, smA, smB);
  int nn = lane & 15, hi = lane >> 4, row0 = row0blk + wave * 16;
#pragma unroll
  for (int t = 0; t < 4; ++t)
#pragma unroll
    for (int e = 0; e < 8; ++e) {
      int row = row0 + e + hi * 8, col = n0 + t * 16 + nn;
      C[(size_t)row * 1024 + col] = acc[t][e] + bout[col];
    }
}

// ---------------- batched GEMMs (direct, register-pipelined) ----------------
__global__ void __launch_bounds__(128) gemm_bf16_f32out(
    const __bf16* __restrict__ A, long long sA, int lda,
    const __bf16* __restrict__ BT, long long sBT, int ldb,
    float* __restrict__ C, long long sC, int ldc, int K) {
  int z = blockIdx.z;
  A += (size_t)z * sA; BT += (size_t)z * sBT; C += (size_t)z * sC;
  int wave = threadIdx.x >> 5, lane = threadIdx.x & 31;
  int row0 = blockIdx.x * 64 + wave * 16, n0 = blockIdx.y * 64;
  v8f acc[4] = {};
  wave_tile_16x64(A, lda, BT, ldb, row0, n0, K, acc);
  int nn = lane & 15, hi = lane >> 4;
#pragma unroll
  for (int t = 0; t < 4; ++t)
#pragma unroll
    for (int e = 0; e < 8; ++e) {
      int row = row0 + e + hi * 8, col = n0 + t * 16 + nn;
      C[(size_t)row * ldc + col] = acc[t][e];
    }
}

// attn_out = attn1 @ Z2 per (b,h); bf16 into [b*4096+i][head*64+j] layout.
__global__ void __launch_bounds__(128) attnout_gemm(const __bf16* __restrict__ attn1,
                                                    const __bf16* __restrict__ Z2T,
                                                    __bf16* __restrict__ out) {
  int z = blockIdx.z;  // bh
  const __bf16* A = attn1 + (size_t)z * 4096 * 64;
  const __bf16* BT = Z2T + (size_t)z * 64 * 64;
  int wave = threadIdx.x >> 5, lane = threadIdx.x & 31;
  int row0 = blockIdx.x * 64 + wave * 16;
  v8f acc[4] = {};
  wave_tile_16x64(A, 64, BT, 64, row0, 0, 64, acc);
  int nn = lane & 15, hi = lane >> 4;
  int bidx = z >> 4, head = z & 15;
#pragma unroll
  for (int t = 0; t < 4; ++t)
#pragma unroll
    for (int e = 0; e < 8; ++e) {
      int i = row0 + e + hi * 8, j = t * 16 + nn;
      out[((size_t)(bidx * 4096 + i)) * 1024 + head * 64 + j] = (__bf16)acc[t][e];
    }
}

// ---------------- softmax ----------------
template <typename T>
__global__ void softmax64_kernel(const float* __restrict__ S, T* __restrict__ out, long long rows) {
  long long row = (long long)blockIdx.x * 8 + (threadIdx.x >> 5);
  if (row >= rows) return;
  int lane = threadIdx.x & 31;
  const float* p = S + row * 64;
  float a = p[lane * 2], b = p[lane * 2 + 1];
  float mx = fmaxf(a, b);
  for (int m = 16; m >= 1; m >>= 1) mx = fmaxf(mx, __shfl_xor(mx, m, 32));
  float ea = __expf(a - mx), eb = __expf(b - mx);
  float s = ea + eb;
  for (int m = 16; m >= 1; m >>= 1) s += __shfl_xor(s, m, 32);
  float inv = 1.f / s;
  out[row * 64 + lane * 2]     = (T)(ea * inv);
  out[row * 64 + lane * 2 + 1] = (T)(eb * inv);
}

__global__ void softmax_wide_kernel(const float* __restrict__ S, __bf16* __restrict__ out, int cols) {
  long long row = blockIdx.x;
  const float* p = S + row * (long long)cols;
  __shared__ float red[256];
  int tid = threadIdx.x;
  float mx = -1e30f;
  for (int c = tid; c < cols; c += 256) mx = fmaxf(mx, p[c]);
  red[tid] = mx; __syncthreads();
  for (int s = 128; s >= 1; s >>= 1) { if (tid < s) red[tid] = fmaxf(red[tid], red[tid + s]); __syncthreads(); }
  mx = red[0]; __syncthreads();
  float sum = 0.f;
  for (int c = tid; c < cols; c += 256) sum += __expf(p[c] - mx);
  red[tid] = sum; __syncthreads();
  for (int s = 128; s >= 1; s >>= 1) { if (tid < s) red[tid] += red[tid + s]; __syncthreads(); }
  float inv = 1.f / red[0];
  for (int c = tid; c < cols; c += 256) out[row * (long long)cols + c] = (__bf16)(__expf(p[c] - mx) * inv);
}

// ---------------- fp32 64x64 matmul in LDS via V_WMMA_F32_16X16X4_F32 ----------------
// So = c0*I + c1*(Sa @ Sb). Safe to alias So with Sa/Sb (regs -> barrier -> write).
DEV void mm64(const float* Sa, const float* Sb, float* So, float c0, float c1) {
  int wave = threadIdx.x >> 5, lane = threadIdx.x & 31;
  int r = lane & 15, hi = lane >> 4, row0 = wave * 16;
  v8f acc[4] = {};
  for (int k0 = 0; k0 < 64; k0 += 4) {
    v2f a;
    a[0] = Sa[(row0 + r) * 64 + k0 + 2 * hi];
    a[1] = Sa[(row0 + r) * 64 + k0 + 2 * hi + 1];
#pragma unroll
    for (int t = 0; t < 4; ++t) {
      v2f b;
      b[0] = Sb[(k0 + 2 * hi) * 64 + t * 16 + r];
      b[1] = Sb[(k0 + 2 * hi + 1) * 64 + t * 16 + r];
      acc[t] = __builtin_amdgcn_wmma_f32_16x16x4_f32(
          false, a, false, b, (short)0, acc[t], false, false);
    }
  }
  __syncthreads();
#pragma unroll
  for (int t = 0; t < 4; ++t)
#pragma unroll
    for (int e = 0; e < 8; ++e) {
      int row = row0 + e + hi * 8, col = t * 16 + r;
      So[row * 64 + col] = (row == col ? c0 : 0.f) + c1 * acc[t][e];
    }
  __syncthreads();
}

// Newton-Schulz pinv: V <- 0.25*V @ (13I - KV @ (15I - KV @ (7I - KV))), 6 iters, fp32.
__global__ void __launch_bounds__(128) pinv_kernel(const float* __restrict__ attn2,
                                                   float* __restrict__ inv) {
  extern __shared__ float sm[];
  float* X = sm; float* V = sm + 4096; float* KV = sm + 8192; float* T = sm + 12288;
  int z = blockIdx.x;
  const float* src = attn2 + (size_t)z * 4096;
  for (int i = threadIdx.x; i < 4096; i += 128) { float v = src[i]; X[i] = v; V[i] = v; }
  __syncthreads();
  for (int it = 0; it < 6; ++it) {
    mm64(X, V, KV, 0.f, 1.f);                                        // KV = X@V
    for (int i = threadIdx.x; i < 4096; i += 128) {
      int rr = i >> 6, cc = i & 63;
      T[i] = (rr == cc ? 7.f : 0.f) - KV[i];                         // T = 7I - KV
    }
    __syncthreads();
    mm64(KV, T, T, 15.f, -1.f);                                      // T = 15I - KV@T
    mm64(KV, T, T, 13.f, -1.f);                                      // T = 13I - KV@T
    mm64(V, T, V, 0.f, 0.25f);                                       // V = 0.25*V@T
  }
  for (int i = threadIdx.x; i < 4096; i += 128) inv[(size_t)z * 4096 + i] = V[i];
}

// Z2 = attn2_inv @ Z (fp32), stored transposed in bf16 for the final batched GEMM.
__global__ void __launch_bounds__(128) z2_kernel(const float* __restrict__ inv,
                                                 const float* __restrict__ Z,
                                                 __bf16* __restrict__ Z2T) {
  extern __shared__ float sm[];
  float* A = sm; float* Bm = sm + 4096; float* O = sm + 8192;
  int z = blockIdx.x;
  for (int i = threadIdx.x; i < 4096; i += 128) {
    A[i] = inv[(size_t)z * 4096 + i];
    Bm[i] = Z[(size_t)z * 4096 + i];
  }
  __syncthreads();
  mm64(A, Bm, O, 0.f, 1.f);
  for (int i = threadIdx.x; i < 4096; i += 128) {
    int rr = i >> 6, cc = i & 63;
    Z2T[(size_t)z * 4096 + cc * 64 + rr] = (__bf16)O[i];
  }
}

// ---------------- host launcher ----------------
extern "C" void kernel_launch(void* const* d_in, const int* in_sizes, int n_in,
                              void* d_out, int out_size, void* d_ws, size_t ws_size,
                              hipStream_t stream) {
  const float* hs   = (const float*)d_in[0];  // [4,4096,1024]
  const float* Wqkv = (const float*)d_in[1];  // [1024,3072]
  const float* bqkv = (const float*)d_in[2];  // [3072]
  const float* Wout = (const float*)d_in[3];  // [1024,1024]
  const float* bout = (const float*)d_in[4];  // [1024]
  float* out = (float*)d_out;                 // [16384,1024]

  char* ws = (char*)d_ws;
  size_t off = 0;
  auto take = [&](size_t bytes) -> char* {
    char* p = ws + off;
    off = (off + bytes + 255) & ~(size_t)255;
    return p;
  };
  __bf16* Xbf   = (__bf16*)take(16777216ull * 2);  // X bf16 [16384,1024]
  __bf16* WqkvT = (__bf16*)take(3145728ull * 2);   // [3072,1024]
  __bf16* WoutT = (__bf16*)take(1048576ull * 2);   // [1024,1024]
  __bf16* qbf   = (__bf16*)take(16777216ull * 2);  // [bh,4096,64] (scaled)
  __bf16* kbf   = (__bf16*)take(16777216ull * 2);  // [bh,4096,64]
  __bf16* vT    = (__bf16*)take(16777216ull * 2);  // [bh,64,4096]
  __bf16* qlbf  = (__bf16*)take(262144ull * 2);    // [bh,64,64]
  __bf16* klbf  = (__bf16*)take(262144ull * 2);    // [bh,64,64]
  float*  Sf    = (float*) take(16777216ull * 4);  // sim scratch (reused)
  __bf16* attn1 = (__bf16*)take(16777216ull * 2);  // [bh,4096,64]
  __bf16* attn3 = (__bf16*)take(16777216ull * 2);  // [bh,64,4096]
  float*  attn2 = (float*) take(262144ull * 4);    // [bh,64,64]
  float*  a2inv = (float*) take(262144ull * 4);
  float*  Zf    = (float*) take(262144ull * 4);    // attn3 @ v
  __bf16* Z2T   = (__bf16*)take(262144ull * 2);    // [bh,64(dh),64(m)]
  __bf16* attnO = (__bf16*)Sf;                     // reuse Sf (dead after softmaxes)

  // 1. precision conversions / pre-transposes
  f32_to_bf16<<<65536, 256, 0, stream>>>(hs, Xbf, 16777216ll);
  transpose_f32_to_bf16<<<12288, 256, 0, stream>>>(Wqkv, WqkvT, 1024, 3072);
  transpose_f32_to_bf16<<<4096, 256, 0, stream>>>(Wout, WoutT, 1024, 1024);

  // 2. QKV projection (103 GFLOP), LDS-staged async double-buffered
  qkv_gemm_staged<<<dim3(128, 48, 1), 256, 0, stream>>>(Xbf, WqkvT, bqkv, qbf, kbf, vT);

  // 3. landmark means
  landmark_means<<<1024, 256, 0, stream>>>(qbf, kbf, qlbf, klbf);

  // 4. sim1 = q @ k_l^T  [bh,4096,64] ; softmax -> attn1 (bf16)
  gemm_bf16_f32out<<<dim3(64, 1, 64), 128, 0, stream>>>(
      qbf, 262144ll, 64, klbf, 4096ll, 64, Sf, 262144ll, 64, 64);
  softmax64_kernel<__bf16><<<32768, 256, 0, stream>>>(Sf, attn1, 262144ll);

  // 5. sim3 = q_l @ k^T  [bh,64,4096] ; softmax over n -> attn3 (bf16)
  gemm_bf16_f32out<<<dim3(1, 64, 64), 128, 0, stream>>>(
      qlbf, 4096ll, 64, kbf, 262144ll, 64, Sf, 262144ll, 4096, 64);
  softmax_wide_kernel<<<4096, 256, 0, stream>>>(Sf, attn3, 4096);

  // 6. sim2 = q_l @ k_l^T ; softmax (fp32, in place)
  gemm_bf16_f32out<<<dim3(1, 1, 64), 128, 0, stream>>>(
      qlbf, 4096ll, 64, klbf, 4096ll, 64, attn2, 4096ll, 64, 64);
  softmax64_kernel<float><<<512, 256, 0, stream>>>(attn2, attn2, 4096ll);

  // 7. Moore-Penrose pinv (fp32 WMMA in LDS)
  pinv_kernel<<<64, 128, 65536, stream>>>(attn2, a2inv);

  // 8. Z = attn3 @ v  [bh,64,64]
  gemm_bf16_f32out<<<dim3(1, 1, 64), 128, 0, stream>>>(
      attn3, 262144ll, 4096, vT, 262144ll, 4096, Zf, 4096ll, 64, 4096);

  // 9. Z2 = attn2_inv @ Z, stored transposed bf16
  z2_kernel<<<64, 128, 49152, stream>>>(a2inv, Zf, Z2T);

  // 10. attn_out = attn1 @ Z2 -> [16384,1024] bf16
  attnout_gemm<<<dim3(64, 1, 64), 128, 0, stream>>>(attn1, Z2T, attnO);

  // 11. out = attn_out @ Wout + bout (34 GFLOP), LDS-staged
  final_gemm_staged<<<dim3(128, 16, 1), 256, 0, stream>>>(attnO, WoutT, bout, out);
}
